// MultiHeadRelativeAttention_89867895701650
// MI455X (gfx1250) — matrix operand order repaired and checked by reading, not compiled, not run on hardware
//
#include <hip/hip_runtime.h>
#include <hip/hip_bf16.h>

// ---------------------------------------------------------------------------
// MI455X (gfx1250) relative multi-head attention, wave32 + WMMA f16->f32.
// Round 3: k-loop unrolled x2 with ping-pong fragment buffers (no v_mov
// buffer rotation), 32x64 per-wave GEMM tiles, partial-wait pipelining.
// ---------------------------------------------------------------------------

typedef __attribute__((ext_vector_type(16))) _Float16 v16h;
typedef __attribute__((ext_vector_type(8)))  float    v8f;

#define S_LEN 1024
#define B_N   4
#define E_DIM 1024
#define H_N   16
#define D_H   64
#define M_POS 2047   // 2*K - 1
#define BAND  1056   // padded band width; cols 0..1038 actually read

// ---- fragment loaders (contiguous k order, same for A and B, wave32) --------------
// A 16x32 f16: lane holds row (lane&15); element i -> K = i + 16*half
static __device__ __forceinline__ v16h ld_a16(const _Float16* base, int ld,
                                              int row, int k0, int halfw) {
  v16h a;
  const _Float16* p = base + (size_t)row * ld + k0 + halfw * 16;
#pragma unroll
  for (int i = 0; i < 8; ++i) a[i] = p[i];
#pragma unroll
  for (int i = 0; i < 8; ++i) a[8 + i] = p[8 + i];
  return a;
}
// B 32x16 f16 where B[k][n] = W[n][k] (i.e. X @ W^T): lane holds col (lane&15);
// element i -> K = i + 16*half  => 16 contiguous halves of W row n.
static __device__ __forceinline__ v16h ld_bT16(const _Float16* base, int ld,
                                               int col, int k0, int halfw) {
  v16h b;
  const _Float16* p = base + (size_t)col * ld + k0 + halfw * 16;
#pragma unroll
  for (int i = 0; i < 16; ++i) b[i] = p[i];
  return b;
}

// ---- f32 -> f16 conversion ------------------------------------------------
__global__ void cvt_f32_f16(const float* __restrict__ src,
                            _Float16* __restrict__ dst, size_t n) {
  size_t i = (size_t)blockIdx.x * blockDim.x + threadIdx.x;
  if (i < n) dst[i] = (_Float16)src[i];
}

// ---- Y = X @ W^T, f16 in / f16 out ----------------------------------------
// block = 4 waves; wave computes a 32x64 tile (2 A strips x 4 B tiles,
// 8 accumulators). k-loop unrolled x2, ping-pong buffers. Requires Kd%64==0.
__global__ __launch_bounds__(128)
void gemm_xwt_f16(const _Float16* __restrict__ X, const _Float16* __restrict__ W,
                  _Float16* __restrict__ Y, int M, int N, int Kd) {
  const int wave = threadIdx.x >> 5;
  const int lane = threadIdx.x & 31;
  const int l16 = lane & 15;
  const int halfw = lane >> 4;
  const int m0 = blockIdx.x * 128 + wave * 32;
  const int n0 = blockIdx.y * 64;
  int arow0 = m0 + l16;        if (arow0 > M - 1) arow0 = M - 1;
  int arow1 = m0 + 16 + l16;   if (arow1 > M - 1) arow1 = M - 1;

  v8f acc[2][4] = {};
  v16h a0[2], b0[4], a1[2], b1[4];
  a0[0] = ld_a16(X, Kd, arow0, 0, halfw);
  a0[1] = ld_a16(X, Kd, arow1, 0, halfw);
#pragma unroll
  for (int t = 0; t < 4; ++t) b0[t] = ld_bT16(W, Kd, n0 + t * 16 + l16, 0, halfw);

  for (int k0 = 0; k0 < Kd; k0 += 64) {
    // stage buf1 for k0+32 (always in range: Kd is a multiple of 64)
    a1[0] = ld_a16(X, Kd, arow0, k0 + 32, halfw);
    a1[1] = ld_a16(X, Kd, arow1, k0 + 32, halfw);
#pragma unroll
    for (int t = 0; t < 4; ++t)
      b1[t] = ld_bT16(W, Kd, n0 + t * 16 + l16, k0 + 32, halfw);
    __builtin_prefetch(X + (size_t)arow0 * Kd + k0 + 64, 0, 3);

    // compute buf0 (k0)
#pragma unroll
    for (int t = 0; t < 4; ++t) {
      acc[0][t] = __builtin_amdgcn_wmma_f32_16x16x32_f16(
          false, a0[0], false, b0[t], (short)0, acc[0][t], false, false);
      acc[1][t] = __builtin_amdgcn_wmma_f32_16x16x32_f16(
          false, a0[1], false, b0[t], (short)0, acc[1][t], false, false);
    }

    // stage buf0 for k0+64 (uniform guard)
    if (k0 + 64 < Kd) {
      a0[0] = ld_a16(X, Kd, arow0, k0 + 64, halfw);
      a0[1] = ld_a16(X, Kd, arow1, k0 + 64, halfw);
#pragma unroll
      for (int t = 0; t < 4; ++t)
        b0[t] = ld_bT16(W, Kd, n0 + t * 16 + l16, k0 + 64, halfw);
    }

    // compute buf1 (k0+32)
#pragma unroll
    for (int t = 0; t < 4; ++t) {
      acc[0][t] = __builtin_amdgcn_wmma_f32_16x16x32_f16(
          false, a1[0], false, b1[t], (short)0, acc[0][t], false, false);
      acc[1][t] = __builtin_amdgcn_wmma_f32_16x16x32_f16(
          false, a1[1], false, b1[t], (short)0, acc[1][t], false, false);
    }
  }
#pragma unroll
  for (int s = 0; s < 2; ++s)
#pragma unroll
    for (int t = 0; t < 4; ++t)
#pragma unroll
      for (int r = 0; r < 8; ++r) {
        int row = m0 + s * 16 + r + 8 * halfw;
        if (row < M)
          Y[(size_t)row * N + n0 + t * 16 + l16] = (_Float16)acc[s][t][r];
      }
}

// ---- Y = X @ W^T + bias, f16 in / f32 out (final projection) --------------
__global__ __launch_bounds__(128)
void gemm_xwt_bias_f32(const _Float16* __restrict__ X, const _Float16* __restrict__ W,
                       const float* __restrict__ bias, float* __restrict__ Y,
                       int M, int N, int Kd) {
  const int wave = threadIdx.x >> 5;
  const int lane = threadIdx.x & 31;
  const int l16 = lane & 15;
  const int halfw = lane >> 4;
  const int m0 = blockIdx.x * 128 + wave * 32;
  const int n0 = blockIdx.y * 64;
  int arow0 = m0 + l16;        if (arow0 > M - 1) arow0 = M - 1;
  int arow1 = m0 + 16 + l16;   if (arow1 > M - 1) arow1 = M - 1;

  v8f acc[2][4] = {};
  v16h a0[2], b0[4], a1[2], b1[4];
  a0[0] = ld_a16(X, Kd, arow0, 0, halfw);
  a0[1] = ld_a16(X, Kd, arow1, 0, halfw);
#pragma unroll
  for (int t = 0; t < 4; ++t) b0[t] = ld_bT16(W, Kd, n0 + t * 16 + l16, 0, halfw);

  for (int k0 = 0; k0 < Kd; k0 += 64) {
    a1[0] = ld_a16(X, Kd, arow0, k0 + 32, halfw);
    a1[1] = ld_a16(X, Kd, arow1, k0 + 32, halfw);
#pragma unroll
    for (int t = 0; t < 4; ++t)
      b1[t] = ld_bT16(W, Kd, n0 + t * 16 + l16, k0 + 32, halfw);

#pragma unroll
    for (int t = 0; t < 4; ++t) {
      acc[0][t] = __builtin_amdgcn_wmma_f32_16x16x32_f16(
          false, a0[0], false, b0[t], (short)0, acc[0][t], false, false);
      acc[1][t] = __builtin_amdgcn_wmma_f32_16x16x32_f16(
          false, a0[1], false, b0[t], (short)0, acc[1][t], false, false);
    }

    if (k0 + 64 < Kd) {
      a0[0] = ld_a16(X, Kd, arow0, k0 + 64, halfw);
      a0[1] = ld_a16(X, Kd, arow1, k0 + 64, halfw);
#pragma unroll
      for (int t = 0; t < 4; ++t)
        b0[t] = ld_bT16(W, Kd, n0 + t * 16 + l16, k0 + 64, halfw);
    }

#pragma unroll
    for (int t = 0; t < 4; ++t) {
      acc[0][t] = __builtin_amdgcn_wmma_f32_16x16x32_f16(
          false, a1[0], false, b1[t], (short)0, acc[0][t], false, false);
      acc[1][t] = __builtin_amdgcn_wmma_f32_16x16x32_f16(
          false, a1[1], false, b1[t], (short)0, acc[1][t], false, false);
    }
  }
#pragma unroll
  for (int s = 0; s < 2; ++s)
#pragma unroll
    for (int t = 0; t < 4; ++t)
#pragma unroll
      for (int r = 0; r < 8; ++r) {
        int row = m0 + s * 16 + r + 8 * halfw;
        int col = n0 + t * 16 + l16;
        if (row < M) Y[(size_t)row * N + col] = acc[s][t][r] + bias[col];
      }
}

// ---- flash-style relative attention ---------------------------------------
// grid = (Q/16, H, B), block = 1 wave. Streams over K, pos term via a
// 16 x BAND pos_raw band computed once per q-tile (rel-shift read from LDS).
__global__ __launch_bounds__(32)
void flash_rel_attn(const _Float16* __restrict__ qh, const _Float16* __restrict__ kh,
                    const _Float16* __restrict__ vh, const _Float16* __restrict__ kph,
                    const float* __restrict__ cb, const float* __restrict__ pb,
                    _Float16* __restrict__ ao) {
  __shared__ _Float16 Pband[16][BAND];
  __shared__ _Float16 probs[16][32];

  const int lane = threadIdx.x & 31;
  const int l16 = lane & 15;
  const int halfw = lane >> 4;
  const int q0 = blockIdx.x * 16;
  const int h  = blockIdx.y;
  const int b  = blockIdx.z;

  // Preload A fragments for q-tile (2 k-steps over D=64), with content / pos
  // bias already folded in (reference adds bias to q before each einsum).
  v16h aC[2], aP[2];
  {
    const int q = q0 + l16;
    const _Float16* qrow = qh + ((size_t)q * B_N + b) * E_DIM + h * D_H;
    const float* cbh = cb + h * D_H;
    const float* pbh = pb + h * D_H;
#pragma unroll
    for (int ks = 0; ks < 2; ++ks)
#pragma unroll
      for (int i = 0; i < 16; ++i) {
        int d = ks * 32 + i + halfw * 16;
        float qv = (float)qrow[d];
        aC[ks][i] = (_Float16)(qv + cbh[d]);
        aP[ks][i] = (_Float16)(qv + pbh[d]);
      }
  }

  // pos_raw band: P[r][c] = sum_d (q+pb)[q0+r][d] * k_pos[mbase+c][d]
  // rel-shift: pos_final[q,j] = pos_raw[q, j-q+K-1]; with mbase = K-1-q0-15
  // the read index becomes c = j + 15 - r  (0..1038).
  const int mbase = (S_LEN - 1) - q0 - 15;  // 1008 - q0
  for (int c0 = 0; c0 < BAND; c0 += 16) {
    v8f pc = {};
    int m = mbase + c0 + l16;
    if (m < 0) m = 0;
    if (m > M_POS - 1) m = M_POS - 1;       // padding cols only; never read
    const _Float16* kpr = kph + (size_t)m * E_DIM + h * D_H;
#pragma unroll
    for (int ks = 0; ks < 2; ++ks) {
      v16h bf;
      const _Float16* p = kpr + ks * 32 + halfw * 16;
#pragma unroll
      for (int i = 0; i < 16; ++i) bf[i] = p[i];
      pc = __builtin_amdgcn_wmma_f32_16x16x32_f16(
          false, aP[ks], false, bf, (short)0, pc, false, false);
    }
#pragma unroll
    for (int r = 0; r < 8; ++r)
      Pband[r + 8 * halfw][c0 + l16] = (_Float16)pc[r];
  }
  __syncthreads();

  const float SCALE = 0.125f * 1.44269504088896f;  // d^-0.5 * log2(e)
  float mrow[8], lrow[8];
#pragma unroll
  for (int r = 0; r < 8; ++r) { mrow[r] = -1e30f; lrow[r] = 0.0f; }
  v8f oacc[4] = {};

  for (int j0 = 0; j0 < S_LEN; j0 += 32) {
    // ---- scores for two 16-wide j subtiles ----
    v8f s[2];
#pragma unroll
    for (int sub = 0; sub < 2; ++sub) {
      v8f sc = {};
      const int j = j0 + sub * 16 + l16;
      const _Float16* krow = kh + ((size_t)j * B_N + b) * E_DIM + h * D_H;
#pragma unroll
      for (int ks = 0; ks < 2; ++ks) {
        v16h bf;
        const _Float16* p = krow + ks * 32 + halfw * 16;
#pragma unroll
        for (int i = 0; i < 16; ++i) bf[i] = p[i];
        sc = __builtin_amdgcn_wmma_f32_16x16x32_f16(
            false, aC[ks], false, bf, (short)0, sc, false, false);
      }
#pragma unroll
      for (int r = 0; r < 8; ++r) {
        int rq = r + 8 * halfw;
        int col = j0 + sub * 16 + l16 + 15 - rq;  // rel-shifted band column
        sc[r] = (sc[r] + (float)Pband[rq][col]) * SCALE;
      }
      s[sub] = sc;
    }

    // ---- online softmax (rows match C-fragment mapping r + 8*half) ----
#pragma unroll
    for (int r = 0; r < 8; ++r) {
      float mx = fmaxf(s[0][r], s[1][r]);
#pragma unroll
      for (int off = 1; off < 16; off <<= 1)
        mx = fmaxf(mx, __shfl_xor(mx, off, 32));
      float mnew = fmaxf(mrow[r], mx);
      float corr = exp2f(mrow[r] - mnew);
      mrow[r] = mnew;
      float p0 = exp2f(s[0][r] - mnew);
      float p1 = exp2f(s[1][r] - mnew);
      float rs = p0 + p1;
#pragma unroll
      for (int off = 1; off < 16; off <<= 1)
        rs += __shfl_xor(rs, off, 32);
      lrow[r] = lrow[r] * corr + rs;
#pragma unroll
      for (int t = 0; t < 4; ++t) oacc[t][r] *= corr;
      int rq = r + 8 * halfw;
      probs[rq][l16]      = (_Float16)p0;
      probs[rq][16 + l16] = (_Float16)p1;
    }
    __syncthreads();

    // ---- probs (16x32) as A fragment, V tile (32x64) as B, 4 WMMAs ----
    v16h ap;
#pragma unroll
    for (int i = 0; i < 16; ++i) {
      int jj = i + 16 * halfw;
      ap[i] = probs[l16][jj];
    }
#pragma unroll
    for (int t = 0; t < 4; ++t) {
      v16h bv;
#pragma unroll
      for (int i = 0; i < 16; ++i) {
        int j = j0 + i + 16 * halfw;
        bv[i] = vh[((size_t)j * B_N + b) * E_DIM + h * D_H + t * 16 + l16];
      }
      oacc[t] = __builtin_amdgcn_wmma_f32_16x16x32_f16(
          false, ap, false, bv, (short)0, oacc[t], false, false);
    }
    __syncthreads();
  }

  // ---- finalize: divide by row sums, write [q*B+b][h*64+d] as f16 ----
#pragma unroll
  for (int t = 0; t < 4; ++t)
#pragma unroll
    for (int r = 0; r < 8; ++r) {
      int rq = r + 8 * halfw;
      int q = q0 + rq;
      float val = oacc[t][r] / lrow[r];
      ao[((size_t)q * B_N + b) * E_DIM + h * D_H + t * 16 + l16] = (_Float16)val;
    }
}

// ---------------------------------------------------------------------------
extern "C" void kernel_launch(void* const* d_in, const int* in_sizes, int n_in,
                              void* d_out, int out_size, void* d_ws, size_t ws_size,
                              hipStream_t stream) {
  const float* query = (const float*)d_in[0];
  const float* key   = (const float*)d_in[1];
  const float* value = (const float*)d_in[2];
  const float* pe    = (const float*)d_in[3];
  const float* w_q   = (const float*)d_in[4];
  const float* w_k   = (const float*)d_in[5];
  const float* w_v   = (const float*)d_in[6];
  const float* w_kp  = (const float*)d_in[7];
  const float* cb    = (const float*)d_in[8];
  const float* pb    = (const float*)d_in[9];
  const float* w_out = (const float*)d_in[10];
  const float* b_out = (const float*)d_in[11];
  float* out = (float*)d_out;

  const size_t ROWS = (size_t)S_LEN * B_N;        // 4096
  const size_t ACT  = ROWS * E_DIM;               // 4M elems
  const size_t WMAT = (size_t)E_DIM * E_DIM;      // 1M elems
  const size_t PEB  = (size_t)M_POS * E_DIM;      // 2047*1024

  char* w = (char*)d_ws;
  _Float16* Xq  = (_Float16*)w; w += ACT  * 2;
  _Float16* Xk  = (_Float16*)w; w += ACT  * 2;
  _Float16* Xv  = (_Float16*)w; w += ACT  * 2;
  _Float16* Wq  = (_Float16*)w; w += WMAT * 2;
  _Float16* Wk  = (_Float16*)w; w += WMAT * 2;
  _Float16* Wv  = (_Float16*)w; w += WMAT * 2;
  _Float16* Wkp = (_Float16*)w; w += WMAT * 2;
  _Float16* Wo  = (_Float16*)w; w += WMAT * 2;
  _Float16* PEh = (_Float16*)w; w += PEB  * 2;
  _Float16* Qp  = (_Float16*)w; w += ACT  * 2;
  _Float16* Kp  = (_Float16*)w; w += ACT  * 2;
  _Float16* Vp  = (_Float16*)w; w += ACT  * 2;
  _Float16* KPp = (_Float16*)w; w += PEB  * 2;
  _Float16* AO  = (_Float16*)w; w += ACT  * 2;

  auto cvt = [&](const float* s, _Float16* d, size_t n) {
    cvt_f32_f16<<<dim3((unsigned)((n + 255) / 256)), dim3(256), 0, stream>>>(s, d, n);
  };
  cvt(query, Xq, ACT);
  cvt(key,   Xk, ACT);
  cvt(value, Xv, ACT);
  cvt(w_q,  Wq,  WMAT);
  cvt(w_k,  Wk,  WMAT);
  cvt(w_v,  Wv,  WMAT);
  cvt(w_kp, Wkp, WMAT);
  cvt(w_out, Wo, WMAT);
  // centered pe slice: lower = PE/2 - K + 1 = 1, length M = 2047 rows
  cvt(pe + (size_t)1 * E_DIM, PEh, PEB);

  // input projections: 4096 x 1024 x 1024 each (block tile 128x64)
  {
    dim3 g(32, 16), blk(128);
    gemm_xwt_f16<<<g, blk, 0, stream>>>(Xq, Wq, Qp, (int)ROWS, E_DIM, E_DIM);
    gemm_xwt_f16<<<g, blk, 0, stream>>>(Xk, Wk, Kp, (int)ROWS, E_DIM, E_DIM);
    gemm_xwt_f16<<<g, blk, 0, stream>>>(Xv, Wv, Vp, (int)ROWS, E_DIM, E_DIM);
  }
  // positional keys: 2047 x 1024 x 1024 (16*128 = 2048 rows, tail clamped)
  gemm_xwt_f16<<<dim3(16, 16), dim3(128), 0, stream>>>(PEh, Wkp, KPp,
                                                       M_POS, E_DIM, E_DIM);
  // attention: one wave per (q-tile, head, batch)
  flash_rel_attn<<<dim3(S_LEN / 16, H_N, B_N), dim3(32), 0, stream>>>(
      Qp, Kp, Vp, KPp, cb, pb, AO);

  // output projection + bias -> f32
  gemm_xwt_bias_f32<<<dim3(32, 16), dim3(128), 0, stream>>>(
      AO, Wo, b_out, out, (int)ROWS, E_DIM, E_DIM);
}
